// RateUnitNetwork_84232898609523
// MI455X (gfx1250) — matrix operand order repaired and checked by reading, not compiled, not run on hardware
//
#include <hip/hip_runtime.h>
#include <hip/hip_bf16.h>
#include <math.h>

// ---------------------------------------------------------------------------
// RateUnitNetwork on MI455X (gfx1250, wave32, WMMA)
//   B=64 T=1000 D=16 H=1024 O=8, a=DT/TAU=0.1
//   h_t = (1-a) h_{t-1} + a (tanh(h_{t-1}) @ W_h2h^T + x_t @ W_i2h^T)
//   out_t = tanh(h_t) @ W_h2o^T   (separate memory-bound post-pass)
// 16 persistent workgroups x 512 threads; one global barrier per timestep.
// Recurrent GEMM via v_wmma_f32_16x16x32_bf16 (f32 accumulate).
// ---------------------------------------------------------------------------

typedef __attribute__((ext_vector_type(16))) __bf16 v16bf;
typedef __attribute__((ext_vector_type(8)))  float  v8f;
typedef __attribute__((ext_vector_type(4)))  float  vf4;

#define B_   64
#define T_   1000
#define D_   16
#define H_   1024
#define O_   8
#define NWG  16
#define LEAK 0.1f

union AB16 { v16bf v; vf4 q[2]; };   // 32 bytes: one WMMA 16-bit operand per lane

// ---- device-scope sense-reversing barrier (16 workgroups) ------------------
__device__ inline void grid_barrier(unsigned* bar) {
  __syncthreads();
  if (threadIdx.x == 0) {
    __threadfence();  // make tanh-buffer stores visible device-wide
    unsigned g = __hip_atomic_load(&bar[16], __ATOMIC_RELAXED, __HIP_MEMORY_SCOPE_AGENT);
    unsigned a = __hip_atomic_fetch_add(&bar[0], 1u, __ATOMIC_ACQ_REL, __HIP_MEMORY_SCOPE_AGENT);
    if (a == NWG - 1u) {
      __hip_atomic_store(&bar[0], 0u, __ATOMIC_RELAXED, __HIP_MEMORY_SCOPE_AGENT);
      __hip_atomic_fetch_add(&bar[16], 1u, __ATOMIC_RELEASE, __HIP_MEMORY_SCOPE_AGENT);
    } else {
      while (__hip_atomic_load(&bar[16], __ATOMIC_ACQUIRE, __HIP_MEMORY_SCOPE_AGENT) == g)
        __builtin_amdgcn_s_sleep(8);
    }
  }
  __syncthreads();
  __threadfence();    // acquire: invalidate near caches before reading peers' tanh
}

// ---- init: pack W_h2h f32 -> bf16 row-major; reset barrier -----------------
__global__ void rnn_init(const float* __restrict__ W_h2h,
                         __bf16* __restrict__ Wpack,
                         unsigned* __restrict__ bar) {
  int i = blockIdx.x * blockDim.x + threadIdx.x;
  if (i < H_ * H_) Wpack[i] = (__bf16)W_h2h[i];
  if (i == 0) { bar[0] = 0u; bar[16] = 0u; }
}

// ---- main recurrent kernel: 16 blocks x 512 threads (16 waves) -------------
__global__ __launch_bounds__(512, 1) void rnn_main(
    const float*  __restrict__ input,    // [B,T,D]
    const float*  __restrict__ hidden0,  // [B,H]
    const float*  __restrict__ W_i2h,    // [H,D]
    const __bf16* __restrict__ Wpack,    // [H,H] bf16 row-major (B operand)
    float*        __restrict__ hidOut,   // [B,T,H] (d_out + 512000)
    __bf16*       __restrict__ bufA,     // 2 x [B,H] bf16 tanh double-buffer
    unsigned*     __restrict__ bar) {
  __shared__ float xs[B_ * D_];          // x_t staging (4 KB)

  const int tid   = threadIdx.x;
  const int wg    = blockIdx.x;          // owns n-columns [wg*64, wg*64+64)
  const int wid   = tid >> 5;            // wave 0..15 -> one 16x16 C tile
  const int lane  = tid & 31;
  const int lo    = lane & 15;
  const int hi    = lane >> 4;
  const int mtile = wid & 3;             // 4 M tiles (B=64)
  const int ntile = wid >> 2;            // 4 N tiles (64 cols per WG)
  const int ng    = wg * 64 + ntile * 16 + lo;   // global n (B/C/D column)

  // W_i2h row for this lane's n (input projection), kept in VGPRs
  float wi[D_];
#pragma unroll
  for (int d = 0; d < D_; ++d) wi[d] = W_i2h[ng * D_ + d];

  // persistent hidden state in the WMMA C/D register layout:
  //   vgpr r, lane -> (m = mtile*16 + 8*hi + r, n = ng)
  float h[8];
#pragma unroll
  for (int r = 0; r < 8; ++r) {
    int mg = mtile * 16 + 8 * hi + r;
    h[r] = hidden0[mg * H_ + ng];
  }

  // prologue: tanh(h0) -> buffer 0 (row-major bf16)
#pragma unroll
  for (int r = 0; r < 8; ++r) {
    int mg = mtile * 16 + 8 * hi + r;
    bufA[mg * H_ + ng] = (__bf16)tanhf(h[r]);
  }
  grid_barrier(bar);

  // A operand row pointer for this lane (ISA 16-bit A layout: m = lane&15,
  // chunks at K = 32k + 8*hi and +16)
  const int aoff = (mtile * 16 + lo) * H_ + 8 * hi;
  // B operand pointer (16-bit B layout: n = lane&15, chunks at K = 32k + 8*hi and +16)
  const __bf16* pb = Wpack + ng * H_ + 8 * hi;

  for (int t = 0; t < T_; ++t) {
    // stage x_t = input[:, t, :] into LDS
    for (int i = tid; i < B_ * D_; i += 512) {
      int b = i >> 4, d = i & 15;
      xs[i] = input[(b * T_ + t) * D_ + d];
    }
    __syncthreads();

    const __bf16* bufR = bufA + (t & 1) * (B_ * H_);
    __bf16*       bufW = bufA + ((t + 1) & 1) * (B_ * H_);
    const __bf16* pa   = bufR + aoff;

    v8f acc = {0.f, 0.f, 0.f, 0.f, 0.f, 0.f, 0.f, 0.f};
#pragma unroll 8
    for (int ks = 0; ks < 32; ++ks) {
      AB16 a, b;
      a.q[0] = *(const vf4*)(pa + 32 * ks);        // K = 32ks+8hi .. +7
      a.q[1] = *(const vf4*)(pa + 32 * ks + 16);   // K = +16 .. +23
      b.q[0] = *(const vf4*)(pb + 32 * ks);        // K = 32ks+8hi .. +7
      b.q[1] = *(const vf4*)(pb + 32 * ks + 16);   // K = +16 .. +23
      acc = __builtin_amdgcn_wmma_f32_16x16x32_bf16(
          false, a.v, false, b.v, (short)0, acc, false, false);
    }

    // h update + stores, directly in C/D layout
#pragma unroll
    for (int r = 0; r < 8; ++r) {
      int mg = mtile * 16 + 8 * hi + r;
      const float* xr = &xs[mg * D_];
      float p = 0.f;
#pragma unroll
      for (int d = 0; d < D_; ++d) p += xr[d] * wi[d];
      float hn = (1.0f - LEAK) * h[r] + LEAK * (acc[r] + p);
      h[r] = hn;
      hidOut[((size_t)mg * T_ + t) * H_ + ng] = hn;       // hiddens output
      bufW[mg * H_ + ng] = (__bf16)tanhf(hn);             // A for step t+1
    }
    grid_barrier(bar);
  }
}

// ---- post-pass: outputs[b,t,o] = tanh(h_t) . W_h2o[o,:]  (memory-bound) ---
__global__ __launch_bounds__(256, 1) void rnn_out(
    const float* __restrict__ hidOut,   // [B,T,H]
    const float* __restrict__ W_h2o,    // [O,H]
    float*       __restrict__ out) {    // [B,T,O]
  __shared__ float wl[O_ * H_];         // 32 KB
  const int tid = threadIdx.x;
  for (int i = tid; i < O_ * H_; i += 256) wl[i] = W_h2o[i];
  __syncthreads();

  const int wid  = tid >> 5;
  const int lane = tid & 31;
  const int row  = blockIdx.x * 8 + wid;        // (b*T + t), one row per wave
  const float* hr = hidOut + (size_t)row * H_;

  float s[O_] = {0.f, 0.f, 0.f, 0.f, 0.f, 0.f, 0.f, 0.f};
  for (int j = 0; j < H_ / 32; ++j) {
    float th = tanhf(hr[lane + 32 * j]);
#pragma unroll
    for (int o = 0; o < O_; ++o) s[o] += th * wl[o * H_ + lane + 32 * j];
  }
#pragma unroll
  for (int o = 0; o < O_; ++o)
#pragma unroll
    for (int m = 16; m >= 1; m >>= 1) s[o] += __shfl_xor(s[o], m, 32);
  if (lane == 0) {
#pragma unroll
    for (int o = 0; o < O_; ++o) out[(size_t)row * O_ + o] = s[o];
  }
}

// ---------------------------------------------------------------------------
extern "C" void kernel_launch(void* const* d_in, const int* in_sizes, int n_in,
                              void* d_out, int out_size, void* d_ws, size_t ws_size,
                              hipStream_t stream) {
  const float* input  = (const float*)d_in[0];   // [64,1000,16]
  const float* hidden = (const float*)d_in[1];   // [64,1024]
  const float* W_i2h  = (const float*)d_in[2];   // [1024,16]
  const float* W_h2h  = (const float*)d_in[3];   // [1024,1024]
  const float* W_h2o  = (const float*)d_in[4];   // [8,1024]

  float* out    = (float*)d_out;                       // outputs [64,1000,8]
  float* hidOut = out + (size_t)B_ * T_ * O_;          // hiddens [64,1000,1024]

  char*     ws    = (char*)d_ws;
  __bf16*   Wpack = (__bf16*)ws;                                 // 2 MB
  __bf16*   bufA  = (__bf16*)(ws + (size_t)2 * 1024 * 1024);     // 256 KB
  unsigned* bar   = (unsigned*)(ws + (size_t)2 * 1024 * 1024 + 256 * 1024);

  rnn_init<<<(H_ * H_ + 255) / 256, 256, 0, stream>>>(W_h2h, Wpack, bar);
  rnn_main<<<NWG, 512, 0, stream>>>(input, hidden, W_i2h, Wpack, hidOut, bufA, bar);
  rnn_out<<<(B_ * T_) / 8, 256, 0, stream>>>(hidOut, W_h2o, out);
}
